// CharacterModel_30554397344036
// MI455X (gfx1250) — matrix operand order repaired and checked by reading, not compiled, not run on hardware
//
#include <hip/hip_runtime.h>
#include <hip/hip_bf16.h>

typedef __attribute__((ext_vector_type(16))) _Float16 v16h;
typedef __attribute__((ext_vector_type(8)))  float    v8f;

union ABFrag { v16h h; unsigned int u[8]; };

#define NWORDS   8192
#define MAXWLEN  16
#define EMBD     64
#define HID      256
#define KDIM     320    // EMB + HID
#define NGATE    1024   // 4*HID
#define KD2      160    // KDIM/2 dwords per fp16 column
#define MTILE    32     // words per block
#define NTHR     512    // 16 waves

__device__ __forceinline__ float fsigmoid(float x) {
    return 1.0f / (1.0f + __expf(-x));
}
__device__ __forceinline__ float ftanh(float x) {
    return 2.0f / (1.0f + __expf(-2.0f * x)) - 1.0f;
}

// Pack [W_ih^T ; W_hh^T] -> fp16, column-major in K: Wc[n*320 + k]
__global__ __launch_bounds__(256) void prep_weights(
    const float* __restrict__ W_ih, const float* __restrict__ W_hh,
    _Float16* __restrict__ Wc)
{
    int idx = blockIdx.x * 256 + threadIdx.x;        // over 1024*320
    if (idx >= NGATE * KDIM) return;
    int n = idx / KDIM, k = idx % KDIM;
    float v = (k < EMBD) ? W_ih[n * EMBD + k] : W_hh[n * HID + (k - EMBD)];
    Wc[n * KDIM + k] = (_Float16)v;
}

__global__ __launch_bounds__(NTHR) void char_lstm_kernel(
    const int* __restrict__ chars, const int* __restrict__ wordlens,
    const int* __restrict__ orig_idx, const float* __restrict__ emb_table,
    const float* __restrict__ b_ih, const float* __restrict__ b_hh,
    const float* __restrict__ attn_w, const float* __restrict__ h_init,
    const float* __restrict__ c_init, const _Float16* __restrict__ Wc,
    float* __restrict__ out)
{
    __shared__ alignas(16) _Float16 Af[MTILE][336]; // [m][k]: k<64 = x_t, 64..319 = h (fp16)
    __shared__ float Hls[MTILE][264];               // h state fp32 (for attention pool)
    __shared__ float bsumS[NGATE];
    __shared__ float attwS[HID];
    __shared__ float part[MTILE][16];
    __shared__ float wlS[MTILE];
    __shared__ int   wlenS[MTILE];

    const int tid   = threadIdx.x;
    const int word0 = blockIdx.x * MTILE;
    const unsigned int* WcU = (const unsigned int*)Wc;

    // ---- block init ----
    for (int i = tid; i < NGATE; i += NTHR) bsumS[i] = b_ih[i] + b_hh[i];
    if (tid < HID) attwS[tid] = attn_w[tid];
    {
        int m = tid >> 4, j = tid & 15;          // 16 threads per word, 32 words
        #pragma unroll
        for (int q = 0; q < 16; ++q) {
            int h = j * 16 + q;
            float hv = h_init[h];
            Hls[m][h] = hv;
            Af[m][EMBD + h] = (_Float16)hv;
        }
    }
    if (tid < MTILE) wlenS[tid] = wordlens[word0 + tid];
    __syncthreads();

    const int lane  = tid & 31;
    const int wv    = tid >> 5;                  // wave id 0..15
    const int grp   = wv >> 3;                   // m-group: 0 -> words 0-15, 1 -> 16-31
    const int w8    = wv & 7;                    // wave-in-group 0..7
    const int mrow0 = grp * 16;                  // base row of this wave's A tile
    const int ln    = lane & 15;
    const int half  = lane >> 4;                 // 0: lanes 0-15, 1: lanes 16-31
    const int m_off = half * 8;                  // C/D row base within the 16-row tile
    const int hcol0 = w8 * 16 + ln;              // this lane's h column (0..127)
    const int hcol1 = 128 + hcol0;               // (128..255)

    float ci0 = c_init[hcol0], ci1 = c_init[hcol1];
    v8f cst0, cst1, res0, res1;
    #pragma unroll
    for (int v = 0; v < 8; ++v) { cst0[v] = ci0; cst1[v] = ci1; res0[v] = 0.f; res1[v] = 0.f; }

    float bias_r[8];
    #pragma unroll
    for (int j = 0; j < 8; ++j) bias_r[j] = bsumS[128 * j + hcol0];

    for (int t = 0; t < MAXWLEN; ++t) {
        // ---- embedding gather: Af[m][0:64] = fp16(emb_table[chars[word,t]]) ----
        {
            int m = tid >> 4, e0 = (tid & 15) * 4;
            int ch = chars[(word0 + m) * MAXWLEN + t];
            const float* er = emb_table + ch * EMBD + e0;
            #pragma unroll
            for (int q = 0; q < 4; ++q) Af[m][e0 + q] = (_Float16)er[q];
        }
        __syncthreads();

        // ---- gates = [x_t|h] @ Wc + b via WMMA f16->f32, K=320 ----
        v8f acc[8] = {};
        const int akb = half ? 8 : 0;
        const int bkb = half ? 8 : 0;
        for (int kk = 0; kk < 10; ++kk) {
            const int kbase = kk * 32;
            ABFrag a;
            #pragma unroll
            for (int j = 0; j < 8; ++j) {        // A layout (ISA 7.12.2, 16-bit 16x32)
                int ko = (j < 4) ? (2 * j + akb) : (16 + 2 * (j - 4) + akb);
                a.u[j] = *(const unsigned int*)&Af[mrow0 + ln][kbase + ko];
            }
            #pragma unroll
            for (int j = 0; j < 8; ++j) {        // ntile = 8*j + w8
                const int nbase = 128 * j + (w8 << 4);
                const unsigned int* col = WcU + (size_t)(nbase + ln) * KD2 + ((kbase + bkb) >> 1);
                ABFrag bf;
                #pragma unroll
                for (int q = 0; q < 8; ++q) bf.u[q] = col[(q < 4) ? q : (q + 4)];
                acc[j] = __builtin_amdgcn_wmma_f32_16x16x32_f16(
                    false, a.h, false, bf.h, (short)0, acc[j], false, false);
            }
        }
        __syncthreads();   // all waves finished reading Af/h before we overwrite

        // ---- LSTM cell update, fully in registers (i,f,g,o = acc[0..7]) ----
        float hn0[8], hn1[8];
        #pragma unroll
        for (int v = 0; v < 8; ++v) {
            int m = mrow0 + m_off + v;
            float ig = fsigmoid(acc[0][v] + bias_r[0]);
            float fg = fsigmoid(acc[2][v] + bias_r[2]);
            float gg = ftanh   (acc[4][v] + bias_r[4]);
            float og = fsigmoid(acc[6][v] + bias_r[6]);
            float cn = fg * cst0[v] + ig * gg;
            cst0[v] = cn;
            float hh = og * ftanh(cn);
            hn0[v] = hh;
            Hls[m][hcol0] = hh;
            Af[m][EMBD + hcol0] = (_Float16)hh;

            ig = fsigmoid(acc[1][v] + bias_r[1]);
            fg = fsigmoid(acc[3][v] + bias_r[3]);
            gg = ftanh   (acc[5][v] + bias_r[5]);
            og = fsigmoid(acc[7][v] + bias_r[7]);
            cn = fg * cst1[v] + ig * gg;
            cst1[v] = cn;
            hh = og * ftanh(cn);
            hn1[v] = hh;
            Hls[m][hcol1] = hh;
            Af[m][EMBD + hcol1] = (_Float16)hh;
        }
        __syncthreads();

        // ---- fused masked sigmoid attention: w[m] = mask * sigmoid(h . attn_w) ----
        {
            int m = tid >> 4, j = tid & 15;
            float s = 0.f;
            #pragma unroll
            for (int q = 0; q < 16; ++q) s += Hls[m][j * 16 + q] * attwS[j * 16 + q];
            part[m][j] = s;
        }
        __syncthreads();
        if (tid < MTILE) {
            float d = 0.f;
            #pragma unroll
            for (int j = 0; j < 16; ++j) d += part[tid][j];
            wlS[tid] = (t < wlenS[tid]) ? fsigmoid(d) : 0.f;
        }
        __syncthreads();
        #pragma unroll
        for (int v = 0; v < 8; ++v) {
            float wm = wlS[mrow0 + m_off + v];
            res0[v] += wm * hn0[v];
            res1[v] += wm * hn1[v];
        }
    }

    // ---- scatter through word_orig_idx (permutation -> full coverage) ----
    #pragma unroll
    for (int v = 0; v < 8; ++v) {
        int m = mrow0 + m_off + v;
        int dst = orig_idx[word0 + m];
        out[(size_t)dst * HID + hcol0] = res0[v];
        out[(size_t)dst * HID + hcol1] = res1[v];
    }
}

extern "C" void kernel_launch(void* const* d_in, const int* in_sizes, int n_in,
                              void* d_out, int out_size, void* d_ws, size_t ws_size,
                              hipStream_t stream)
{
    const int*   chars   = (const int*)d_in[0];
    const int*   wlens   = (const int*)d_in[1];
    const int*   origidx = (const int*)d_in[2];
    const float* emb     = (const float*)d_in[3];
    const float* W_ih    = (const float*)d_in[4];
    const float* W_hh    = (const float*)d_in[5];
    const float* b_ih    = (const float*)d_in[6];
    const float* b_hh    = (const float*)d_in[7];
    const float* attn_w  = (const float*)d_in[8];
    const float* h_init  = (const float*)d_in[9];
    const float* c_init  = (const float*)d_in[10];
    float* out = (float*)d_out;
    _Float16* Wc = (_Float16*)d_ws;   // 1024*320 fp16 = 640 KB

    prep_weights<<<(NGATE * KDIM + 255) / 256, 256, 0, stream>>>(W_ih, W_hh, Wc);
    char_lstm_kernel<<<NWORDS / MTILE, NTHR, 0, stream>>>(
        chars, wlens, origidx, emb, b_ih, b_hh, attn_w, h_init, c_init, Wc, out);
}
